// GraphEncoderAdaptExtraFeatures_56968446214432
// MI455X (gfx1250) — matrix operand order repaired and checked by reading, not compiled, not run on hardware
//
#include <hip/hip_runtime.h>

// ---------------------------------------------------------------------------
// Types for CDNA5 WMMA (wave32, v_wmma_f32_16x16x32_bf16)
// ---------------------------------------------------------------------------
typedef __attribute__((ext_vector_type(16))) __bf16 v16bf;
typedef __attribute__((ext_vector_type(8)))  float  v8f;
typedef unsigned short u16;

__device__ __forceinline__ u16 f2bf(float f) {
  unsigned u = __float_as_uint(f);
  unsigned r = u + 0x7FFFu + ((u >> 16) & 1u);   // round-to-nearest-even
  return (u16)(r >> 16);
}
__device__ __forceinline__ float bf2f(u16 h) {
  return __uint_as_float(((unsigned)h) << 16);
}

// Inverse map of the CDNA5 16-bit A-matrix 16x32 fragment layout:
// element (row m, k) -> LDS index  kt*512 + (half*16+m)*16 + j   (u16 units)
// K(p,sub,half) = ((p&3)*2+sub) + half*8 + (p>>2)*16  (per ISA 7.12.2)
__device__ __forceinline__ int afrag_index(int m, int k) {
  int kt   = k >> 5;
  int kk   = k & 31;
  int half = (kk >> 3) & 1;
  int p    = ((kk & 7) >> 1) + ((kk >> 4) << 2);
  int j    = (p << 1) + (kk & 1);
  return kt * 512 + (half * 16 + m) * 16 + j;
}

// ---------------------------------------------------------------------------
// k_zero: grid-stride fp32 zero fill
// ---------------------------------------------------------------------------
__global__ void k_zero(float* __restrict__ p, int n) {
  for (int i = blockIdx.x * blockDim.x + threadIdx.x; i < n;
       i += gridDim.x * blockDim.x)
    p[i] = 0.0f;
}

// ---------------------------------------------------------------------------
// k_deg: degree histogram of dst (scatter-mean denominator), computed once
// ---------------------------------------------------------------------------
__global__ void k_deg(const int* __restrict__ dst, float* __restrict__ deg, int E) {
  int e = blockIdx.x * blockDim.x + threadIdx.x;
  if (e < E) atomicAdd(&deg[dst[e]], 1.0f);
}

// ---------------------------------------------------------------------------
// k_embed: NeRF positional encoding -> h0 [N, 224] bf16 (200 real + 24 pad)
// layout per block b(0..2): [v(3), f-major (sin3,cos3)*10] then x[:,9:20]
// ---------------------------------------------------------------------------
__global__ void k_embed(const float* __restrict__ x, u16* __restrict__ hbf, int N) {
  const int KP = 224;
  int idx = blockIdx.x * blockDim.x + threadIdx.x;
  int total = N * KP;
  if (idx >= total) return;
  int n = idx / KP, j = idx % KP;
  float val = 0.0f;
  if (j < 189) {
    int blk = j / 63, w = j % 63;
    if (w < 3) {
      val = x[n * 20 + blk * 3 + w];
    } else {
      int t = w - 3;
      int f = t / 6, rem = t % 6;
      int s = rem / 3, comp = rem % 3;
      float ang = x[n * 20 + blk * 3 + comp] * (float)(1 << f);
      val = s ? __cosf(ang) : __sinf(ang);
    }
  } else if (j < 200) {
    val = x[n * 20 + 9 + (j - 189)];
  }
  hbf[n * KP + j] = f2bf(val);
}

// ---------------------------------------------------------------------------
// k_packB: pack an fp32 row-major [K x Ncols] matrix into bf16 B-fragment
// order for v_wmma_f32_16x16x32_bf16.
//   frag index fi = ((kt*ntiles + nt)*32 + lane)*16 + j
//   element: K = (j&7) + 8*(lane>>4) + 16*(j>>3) + 32*kt, col = nt*16 + (lane&15)
// mode 0: src[k][c]          (W2)
// mode 1: src[k][c] - src[k+Ksrc][c]   (W1a - W1b, b1-folded path)
// mode 2: src[k+Ksrc][c]               (W1b)
// Rows k >= Ksrc are zero (K padding).
// ---------------------------------------------------------------------------
__global__ void k_packB(const float* __restrict__ src, u16* __restrict__ dst,
                        int Kpad, int Ksrc, int Ncols, int mode) {
  int fi = blockIdx.x * blockDim.x + threadIdx.x;
  int total = Kpad * Ncols;
  if (fi >= total) return;
  int ntiles = Ncols >> 4;
  int j = fi & 15, lane = (fi >> 4) & 31, rest = fi >> 9;
  int nt = rest % ntiles, kt = rest / ntiles;
  int kg = (j & 7) + ((lane >> 4) << 3) + ((j >> 3) << 4) + (kt << 5);
  int col = (nt << 4) + (lane & 15);
  float v = 0.0f;
  if (kg < Ksrc) {
    if (mode == 0)      v = src[kg * Ncols + col];
    else if (mode == 1) v = src[kg * Ncols + col] - src[(kg + Ksrc) * Ncols + col];
    else                v = src[(kg + Ksrc) * Ncols + col];
  }
  dst[fi] = f2bf(v);
}

// ---------------------------------------------------------------------------
// k_node_gemm: out[n,c] = sum_k hbf[n,k] * wpack[k,c] (+ bias[c]), bf16 out.
// One wave per 16-node tile. A staged in LDS in fragment order, then
// ntiles x ktiles WMMA loop.
// ---------------------------------------------------------------------------
__global__ void k_node_gemm(const u16* __restrict__ hbf,
                            const u16* __restrict__ wpack,
                            const float* __restrict__ bias,
                            u16* __restrict__ outPQ,
                            int N, int Kpad, int dout) {
  __shared__ __attribute__((aligned(32))) u16 ldsA[16 * 256]; // Kpad <= 256
  const int lane = threadIdx.x;
  const int n0 = blockIdx.x * 16;

  // stage 16 rows of hbf into A-fragment order (coalesced over k)
  for (int t = lane; t < 16 * Kpad; t += 32) {
    int m = t / Kpad, k = t % Kpad;
    int n = n0 + m;
    u16 v = (n < N) ? hbf[n * Kpad + k] : (u16)0;
    ldsA[afrag_index(m, k)] = v;
  }
  __syncthreads();

  const int ntiles = dout >> 4, ktiles = Kpad >> 5;
  const int half = lane >> 4, lcol = lane & 15;
  for (int nt = 0; nt < ntiles; nt++) {
    float bv = bias ? bias[(nt << 4) + lcol] : 0.0f;
    v8f c;
#pragma unroll
    for (int i = 0; i < 8; i++) c[i] = bv;
    for (int kt = 0; kt < ktiles; kt++) {
      v16bf a = *(const v16bf*)(ldsA + kt * 512 + lane * 16);
      v16bf b = *(const v16bf*)(wpack + (((kt * ntiles + nt) * 32 + lane) << 4));
      c = __builtin_amdgcn_wmma_f32_16x16x32_bf16(false, a, false, b,
                                                  (short)0, c, false, false);
    }
#pragma unroll
    for (int i = 0; i < 8; i++) {
      int row = half * 8 + i;
      int n = n0 + row;
      if (n < N) outPQ[n * dout + (nt << 4) + lcol] = f2bf(c[i]);
    }
  }
}

// ---------------------------------------------------------------------------
// k_edge_gemm: per 16-edge tile (one wave):
//   pre[m,k] = relu(P[dst[m],k] + Q[src[m],k])   (b1 folded into P)
//   out      = pre @ W2 (+ b2), atomic scatter-add into acc[dst[m]]
// ---------------------------------------------------------------------------
__global__ void k_edge_gemm(const int* __restrict__ srcI,
                            const int* __restrict__ dstI,
                            const u16* __restrict__ P,
                            const u16* __restrict__ Q,
                            const u16* __restrict__ w2p,
                            const float* __restrict__ b2,
                            float* __restrict__ acc,
                            int E, int dout) {
  __shared__ __attribute__((aligned(32))) u16 ldsA[16 * 512]; // dout <= 512
  __shared__ int sIdx[16], dIdx[16];
  const int lane = threadIdx.x;
  const int base = blockIdx.x * 16;

  if (lane < 16) {
    int e = base + lane;
    sIdx[lane] = (e < E) ? srcI[e] : 0;
    dIdx[lane] = (e < E) ? dstI[e] : -1;
  }
  __syncthreads();

  // stage relu(P[dst]+Q[src]) as bf16 in A-fragment order; coalesced over k
  for (int t = lane; t < 16 * dout; t += 32) {
    int m = t / dout, k = t % dout;
    int d = dIdx[m]; int dd = d < 0 ? 0 : d;
    int s = sIdx[m];
    float v = bf2f(P[dd * dout + k]) + bf2f(Q[s * dout + k]);
    v = fmaxf(v, 0.0f);
    ldsA[afrag_index(m, k)] = f2bf(v);
  }
  __syncthreads();

  const int ntiles = dout >> 4, ktiles = dout >> 5;
  const int half = lane >> 4, lcol = lane & 15;
  for (int nt = 0; nt < ntiles; nt++) {
    v8f c;
#pragma unroll
    for (int i = 0; i < 8; i++) c[i] = 0.0f;
    for (int kt = 0; kt < ktiles; kt++) {
      v16bf a = *(const v16bf*)(ldsA + kt * 512 + lane * 16);
      v16bf b = *(const v16bf*)(w2p + (((kt * ntiles + nt) * 32 + lane) << 4));
      c = __builtin_amdgcn_wmma_f32_16x16x32_bf16(false, a, false, b,
                                                  (short)0, c, false, false);
    }
    int col = (nt << 4) + lcol;
    float bv = b2[col];
#pragma unroll
    for (int i = 0; i < 8; i++) {
      int row = half * 8 + i;
      int d = dIdx[row];
      if (d >= 0) atomicAdd(&acc[d * dout + col], c[i] + bv);
    }
  }
}

// ---------------------------------------------------------------------------
// k_bn_stats: per-channel sum / sumsq of acc/max(deg,1). blockDim.x == dout.
// ---------------------------------------------------------------------------
__global__ void k_bn_stats(const float* __restrict__ acc,
                           const float* __restrict__ deg,
                           float* __restrict__ sum, float* __restrict__ sumsq,
                           int N, int dout, int rowsPerBlock) {
  int c = threadIdx.x;
  int n0 = blockIdx.x * rowsPerBlock;
  float s = 0.0f, q = 0.0f;
  for (int r = 0; r < rowsPerBlock; r++) {
    int n = n0 + r;
    if (n >= N) break;
    float v = acc[n * dout + c] / fmaxf(deg[n], 1.0f);
    s += v; q += v * v;
  }
  atomicAdd(&sum[c], s);
  atomicAdd(&sumsq[c], q);
}

// ---------------------------------------------------------------------------
// k_bn_norm: h = relu(g*(v-mu)*rsqrt(var+1e-5)+b) -> bf16 next-layer input
// ---------------------------------------------------------------------------
__global__ void k_bn_norm(const float* __restrict__ acc,
                          const float* __restrict__ deg,
                          const float* __restrict__ sum,
                          const float* __restrict__ sumsq,
                          const float* __restrict__ g,
                          const float* __restrict__ b,
                          u16* __restrict__ hbf, int N, int dout) {
  int idx = blockIdx.x * blockDim.x + threadIdx.x;
  if (idx >= N * dout) return;
  int n = idx / dout, c = idx % dout;
  float v  = acc[idx] / fmaxf(deg[n], 1.0f);
  float invN = 1.0f / (float)N;
  float mu = sum[c] * invN;
  float var = sumsq[c] * invN - mu * mu;
  float o = g[c] * (v - mu) * rsqrtf(var + 1e-5f) + b[c];
  hbf[idx] = f2bf(fmaxf(o, 0.0f));
}

// ---------------------------------------------------------------------------
// k_final: out = acc / max(deg,1)  (layer 5, fp32 output)
// ---------------------------------------------------------------------------
__global__ void k_final(const float* __restrict__ acc,
                        const float* __restrict__ deg,
                        float* __restrict__ out, int N, int dout) {
  int idx = blockIdx.x * blockDim.x + threadIdx.x;
  if (idx >= N * dout) return;
  int n = idx / dout;
  out[idx] = acc[idx] / fmaxf(deg[n], 1.0f);
}

// ---------------------------------------------------------------------------
// Host orchestration
// ---------------------------------------------------------------------------
extern "C" void kernel_launch(void* const* d_in, const int* in_sizes, int n_in,
                              void* d_out, int out_size, void* d_ws, size_t ws_size,
                              hipStream_t stream) {
  (void)n_in; (void)out_size; (void)ws_size;
  const float* x  = (const float*)d_in[0];
  const int*   ei = (const int*)d_in[1];
  const int N = in_sizes[2];
  const int E = in_sizes[1] / 2;
  const int* srcI = ei;
  const int* dstI = ei + E;

  const float *gw1[5], *gb1[5], *gw2[5], *gb2[5];
  for (int l = 0; l < 5; l++) {
    gw1[l] = (const float*)d_in[3 + l * 4 + 0];
    gb1[l] = (const float*)d_in[3 + l * 4 + 1];
    gw2[l] = (const float*)d_in[3 + l * 4 + 2];
    gb2[l] = (const float*)d_in[3 + l * 4 + 3];
  }
  const float *ng[4], *nb[4];
  for (int i = 0; i < 4; i++) {
    ng[i] = (const float*)d_in[23 + 2 * i];
    nb[i] = (const float*)d_in[24 + 2 * i];
  }

  const int dinA[5]  = {200, 64, 128, 256, 256};
  const int KpadA[5] = {224, 64, 128, 256, 256};
  const int doutA[5] = {64, 128, 256, 256, 512};

  // workspace carve (256B aligned)
  char* w = (char*)d_ws;
  auto carve = [&](size_t bytes) -> void* {
    void* p = (void*)w;
    w += (bytes + 255) & ~(size_t)255;
    return p;
  };
  u16*   hbf   = (u16*)  carve((size_t)N * 256 * 2);
  u16*   Pbf   = (u16*)  carve((size_t)N * 512 * 2);
  u16*   Qbf   = (u16*)  carve((size_t)N * 512 * 2);
  float* acc   = (float*)carve((size_t)N * 512 * 4);
  float* deg   = (float*)carve((size_t)N * 4);
  u16*   w1d   = (u16*)  carve((size_t)256 * 512 * 2);
  u16*   w1b   = (u16*)  carve((size_t)256 * 512 * 2);
  u16*   w2p   = (u16*)  carve((size_t)512 * 512 * 2);
  float* bnsum = (float*)carve(512 * 4);
  float* bnsq  = (float*)carve(512 * 4);

  auto zgrid = [](int n) { int g = (n + 255) / 256; return g > 4096 ? 4096 : g; };

  // degrees (same for every layer)
  k_zero<<<zgrid(N), 256, 0, stream>>>(deg, N);
  k_deg<<<(E + 255) / 256, 256, 0, stream>>>(dstI, deg, E);

  // embedding -> h0 (bf16, stride 224)
  {
    int total = N * 224;
    k_embed<<<(total + 255) / 256, 256, 0, stream>>>(x, hbf, N);
  }

  for (int l = 0; l < 5; l++) {
    const int din = dinA[l], Kpad = KpadA[l], dout = doutA[l];
    const int packN = Kpad * dout;
    const int packW2 = dout * dout;

    // pack weights into B-fragment order
    k_packB<<<(packN + 255) / 256, 256, 0, stream>>>(gw1[l], w1d, Kpad, din, dout, 1);
    k_packB<<<(packN + 255) / 256, 256, 0, stream>>>(gw1[l], w1b, Kpad, din, dout, 2);
    k_packB<<<(packW2 + 255) / 256, 256, 0, stream>>>(gw2[l], w2p, dout, dout, dout, 0);

    // node-level GEMMs: P = h*(W1a-W1b)+b1, Q = h*W1b  (bf16, WMMA)
    int ntileBlocks = (N + 15) / 16;
    k_node_gemm<<<ntileBlocks, 32, 0, stream>>>(hbf, w1d, gb1[l], Pbf, N, Kpad, dout);
    k_node_gemm<<<ntileBlocks, 32, 0, stream>>>(hbf, w1b, nullptr, Qbf, N, Kpad, dout);

    // per-edge GEMM + scatter-add
    k_zero<<<zgrid(N * dout), 256, 0, stream>>>(acc, N * dout);
    int etileBlocks = (E + 15) / 16;
    k_edge_gemm<<<etileBlocks, 32, 0, stream>>>(srcI, dstI, Pbf, Qbf, w2p, gb2[l],
                                                acc, E, dout);

    if (l < 4) {
      // fused scatter-mean + BatchNorm + ReLU -> next-layer bf16 input
      k_zero<<<1, 256, 0, stream>>>(bnsum, dout);
      k_zero<<<1, 256, 0, stream>>>(bnsq, dout);
      int rows = 64;
      k_bn_stats<<<(N + rows - 1) / rows, dout, 0, stream>>>(acc, deg, bnsum, bnsq,
                                                             N, dout, rows);
      int total = N * dout;
      k_bn_norm<<<(total + 255) / 256, 256, 0, stream>>>(acc, deg, bnsum, bnsq,
                                                         ng[l], nb[l], hbf, N, dout);
    } else {
      int total = N * dout;
      k_final<<<(total + 255) / 256, 256, 0, stream>>>(acc, deg, (float*)d_out,
                                                       N, dout);
    }
  }
}
